// GraphProcessor_2070174236991
// MI455X (gfx1250) — matrix operand order repaired and checked by reading, not compiled, not run on hardware
//
#include <hip/hip_runtime.h>

#define DD      128
#define NNODES  20000
#define NEDGES  80000
#define MPITERS 15
#define TILE    64
#define THREADS 128
#define HP      136      // bf16 LDS activation pitch (elements)
#define LP      132      // f32  LDS layernorm pitch (elements)
#define LN_EPS  1e-5f

typedef __bf16 bf16_t;
typedef __attribute__((ext_vector_type(8)))  __bf16 v8bf;
typedef __attribute__((ext_vector_type(16))) __bf16 v16bf;
typedef __attribute__((ext_vector_type(8)))  float  v8f;
typedef __attribute__((ext_vector_type(4)))  float  v4f;
typedef __attribute__((ext_vector_type(4)))  unsigned int u32x4;
typedef __attribute__((ext_vector_type(8)))  int    i32x8;
typedef __attribute__((ext_vector_type(4)))  int    i32x4;

#define WMMA_BF16(a, b, c) \
  __builtin_amdgcn_wmma_f32_16x16x32_bf16(false, (a), false, (b), (short)0, (c), false, false)

static __device__ inline v8f vzero8() {
  v8f z;
#pragma unroll
  for (int i = 0; i < 8; ++i) z[i] = 0.0f;
  return z;
}

// ---- Tensor Data Mover: 1-D f32 tile, global -> LDS ----------------------
static __device__ inline void tdm_load_f32(const float* gsrc, void* ldsDst, int nElems) {
  unsigned long long ga = (unsigned long long)(size_t)gsrc;
  unsigned ldsOff = (unsigned)(size_t)ldsDst;          // low 32 bits = LDS offset
  u32x4 g0;
  g0[0] = 1u;                                           // count=1 (valid user D#)
  g0[1] = ldsOff;                                       // lds_addr
  g0[2] = (unsigned)(ga & 0xffffffffu);                 // global_addr[31:0]
  g0[3] = (unsigned)((ga >> 32) & 0x01ffffffu) | 0x80000000u;  // addr[56:32] | type=2
  i32x8 g1;
  g1[0] = 2 << 16;                                      // data_size = 4 bytes
  g1[1] = (nElems & 0xffff) << 16;                      // tensor_dim0[15:0]
  g1[2] = ((nElems >> 16) & 0xffff) | (1 << 16);        // dim0 hi | tensor_dim1=1
  g1[3] = (nElems & 0xffff) << 16;                      // tile_dim0
  g1[4] = 1;                                            // tile_dim1=1, tile_dim2=0
  g1[5] = nElems;                                       // tensor_dim0_stride lo
  g1[6] = 0;
  g1[7] = 0;
  i32x4 z4 = {0, 0, 0, 0};
#if defined(__clang_major__) && (__clang_major__ >= 23)
  i32x8 z8 = {0, 0, 0, 0, 0, 0, 0, 0};
  __builtin_amdgcn_tensor_load_to_lds(g0, g1, z4, z4, z8, 0);
#else
  __builtin_amdgcn_tensor_load_to_lds(g0, g1, z4, z4, 0);
#endif
}

// A fragment (16x32 bf16, this lane's share) built from f32 global data.
static __device__ inline v16bf afrag_f32(const float* __restrict__ p) {
  v4f c0 = *(const v4f*)(p);
  v4f c1 = *(const v4f*)(p + 4);
  v4f c2 = *(const v4f*)(p + 16);
  v4f c3 = *(const v4f*)(p + 20);
  v16bf a;
#pragma unroll
  for (int i = 0; i < 4; ++i) {
    a[i]      = (bf16_t)c0[i];
    a[4 + i]  = (bf16_t)c1[i];
    a[8 + i]  = (bf16_t)c2[i];
    a[12 + i] = (bf16_t)c3[i];
  }
  return a;
}

static __device__ inline v16bf afrag_lds(const bf16_t* p) {
  v8bf c0 = *(const v8bf*)(p);
  v8bf c1 = *(const v8bf*)(p + 16);
  v16bf a;
#pragma unroll
  for (int i = 0; i < 8; ++i) { a[i] = c0[i]; a[8 + i] = c1[i]; }
  return a;
}

// B fragment (32x16 bf16): lane = output column n, K values p[0..7] and p[16..23].
static __device__ inline v16bf bfrag(const bf16_t* __restrict__ p) {
  v8bf c0 = *(const v8bf*)(p);
  v8bf c1 = *(const v8bf*)(p + 16);
  v16bf b;
#pragma unroll
  for (int i = 0; i < 8; ++i) { b[i] = c0[i]; b[8 + i] = c1[i]; }
  return b;
}

// 128-K GEMM layer: A from LDS, B from global (batched loads -> WMMA burst).
static __device__ inline void gemm128(const bf16_t* arow, const bf16_t* __restrict__ wt,
                                      int m16, int half, v8f acc[8]) {
#pragma unroll
  for (int kk = 0; kk < 4; ++kk) {
    v16bf b[8];
#pragma unroll
    for (int t = 0; t < 8; ++t)
      b[t] = bfrag(wt + (size_t)(t * 16 + m16) * DD + kk * 32 + half * 8);
    v16bf a = afrag_lds(arow + kk * 32 + half * 8);
#pragma unroll
    for (int t = 0; t < 8; ++t) acc[t] = WMMA_BF16(a, b[t], acc[t]);
  }
}

// acc (+bias, ReLU) -> bf16 LDS activation buffer, then clear acc.
static __device__ inline void bias_relu_store(bf16_t* Hbuf, const float* bias,
                                              int m16, int mbase, v8f acc[8]) {
#pragma unroll
  for (int t = 0; t < 8; ++t) {
    const int n = t * 16 + m16;
    const float bb = bias[n];
#pragma unroll
    for (int j = 0; j < 8; ++j) {
      float v = acc[t][j] + bb;
      Hbuf[(mbase + j) * HP + n] = (bf16_t)(v > 0.0f ? v : 0.0f);
    }
    acc[t] = vzero8();
  }
}

static __device__ inline void bias_store_f32(float* LNbuf, const float* bias,
                                             int m16, int mbase, v8f acc[8]) {
#pragma unroll
  for (int t = 0; t < 8; ++t) {
    const int n = t * 16 + m16;
    const float bb = bias[n];
#pragma unroll
    for (int j = 0; j < 8; ++j)
      LNbuf[(mbase + j) * LP + n] = acc[t][j] + bb;
  }
}

// ---------------------------------------------------------------------------
// Edge kernel: e' = LN(MLP(cat(x[row], x[col], e))) + e ; agg[col] += e'
// params = packed [b0 | b1 | b2 | gamma | beta], 640 f32 (loaded via TDM)
// ---------------------------------------------------------------------------
__global__ __launch_bounds__(THREADS) void edge_kernel(
    const float* __restrict__ xbuf, float* __restrict__ ebuf,
    float* __restrict__ agg, const int* __restrict__ eidx,
    const bf16_t* __restrict__ w0t, const bf16_t* __restrict__ w1t,
    const bf16_t* __restrict__ w2t, const float* __restrict__ params) {
  __shared__ __align__(16) unsigned char smem[TILE * LP * 4];  // Hbuf/LNbuf union
  __shared__ __align__(16) float Pbuf[5 * DD];
  bf16_t* Hbuf  = (bf16_t*)smem;
  float*  LNbuf = (float*)smem;

  const int tid  = threadIdx.x;
  const int wave = tid >> 5;
  const int lane = tid & 31;
  const int half = lane >> 4;
  const int m16  = lane & 15;
  const int rowT = wave * 16 + m16;
  const int edge = blockIdx.x * TILE + rowT;        // 80000 = 1250*64, exact
  const int rsrc = eidx[edge];
  const int csrc = eidx[NEDGES + edge];

  // TDM: DMA the parameter block into LDS (wave 0); prefetch later weights.
  if (wave == 0) {
    tdm_load_f32(params, Pbuf, 5 * DD);
    __builtin_amdgcn_s_wait_tensorcnt(0);
  } else if (wave == 1) {
#pragma unroll
    for (int i = 0; i < 4; ++i)
      __builtin_prefetch(w1t + ((size_t)(i * 32 + lane)) * 128, 0, 3);
  } else if (wave == 2) {
#pragma unroll
    for (int i = 0; i < 4; ++i)
      __builtin_prefetch(w2t + ((size_t)(i * 32 + lane)) * 128, 0, 3);
  }
  __syncthreads();

  const float* s0 = xbuf + (size_t)rsrc * DD;
  const float* s1 = xbuf + (size_t)csrc * DD;
  const float* s2 = ebuf + (size_t)edge * DD;

  v8f acc[8];
#pragma unroll
  for (int t = 0; t < 8; ++t) acc[t] = vzero8();

  // ---- Layer 1: [64,384] x [384,128], A gathered from global f32 ----
  for (int kk = 0; kk < 12; ++kk) {
    const int kwin = kk * 32;
    const float* s = (kwin < DD)     ? (s0 + kwin)
                   : (kwin < 2 * DD) ? (s1 + (kwin - DD))
                                     : (s2 + (kwin - 2 * DD));
    v16bf b[8];
#pragma unroll
    for (int t = 0; t < 8; ++t)
      b[t] = bfrag(w0t + (size_t)(t * 16 + m16) * 384 + kwin + half * 8);
    v16bf a = afrag_f32(s + half * 8);
#pragma unroll
    for (int t = 0; t < 8; ++t) acc[t] = WMMA_BF16(a, b[t], acc[t]);
  }
  const int mbase = wave * 16 + half * 8;
  bias_relu_store(Hbuf, Pbuf, m16, mbase, acc);

  // ---- Layers 2 & 3 (per-wave rows; DS ops are in-order per wave) ----
  const bf16_t* arow = Hbuf + (size_t)rowT * HP;
  gemm128(arow, w1t, m16, half, acc);
  bias_relu_store(Hbuf, Pbuf + DD, m16, mbase, acc);
  gemm128(arow, w2t, m16, half, acc);
  __syncthreads();                                   // Hbuf reads done -> reuse as LNbuf
  bias_store_f32(LNbuf, Pbuf + 2 * DD, m16, mbase, acc);

  // ---- LayerNorm + residual + scatter (wave-local rows) ----
  const float* gam = Pbuf + 3 * DD;
  const float* bet = Pbuf + 4 * DD;
  const float* hrow = LNbuf + (size_t)rowT * LP;
  float sum = 0.0f, sq = 0.0f;
  for (int n4 = 0; n4 < DD; n4 += 4) {
    v4f h = *(const v4f*)(hrow + n4);
#pragma unroll
    for (int j = 0; j < 4; ++j) { sum += h[j]; sq += h[j] * h[j]; }
  }
  const float mu  = sum * (1.0f / DD);
  const float var = sq * (1.0f / DD) - mu * mu;
  const float rs  = rsqrtf(var + LN_EPS);

  float* eout = ebuf + (size_t)edge * DD;
  float* aggp = agg + (size_t)csrc * DD;
  const int nb = half * 64;
  for (int n4 = 0; n4 < 64; n4 += 4) {
    const int n = nb + n4;
    v4f h  = *(const v4f*)(hrow + n);
    v4f eo = *(const v4f*)(eout + n);
    v4f y;
#pragma unroll
    for (int j = 0; j < 4; ++j)
      y[j] = gam[n + j] * (h[j] - mu) * rs + bet[n + j] + eo[j];
    *(v4f*)(eout + n) = y;
#pragma unroll
    for (int j = 0; j < 4; ++j) atomicAdd(&aggp[n + j], y[j]);
  }
}

// ---------------------------------------------------------------------------
// Node kernel: x' = LN(MLP(cat(x, agg))) + x
// ---------------------------------------------------------------------------
__global__ __launch_bounds__(THREADS) void node_kernel(
    float* __restrict__ xbuf, const float* __restrict__ agg,
    const bf16_t* __restrict__ w0t, const bf16_t* __restrict__ w1t,
    const bf16_t* __restrict__ w2t, const float* __restrict__ params) {
  __shared__ __align__(16) unsigned char smem[TILE * LP * 4];
  __shared__ __align__(16) float Pbuf[5 * DD];
  bf16_t* Hbuf  = (bf16_t*)smem;
  float*  LNbuf = (float*)smem;

  const int tid  = threadIdx.x;
  const int wave = tid >> 5;
  const int lane = tid & 31;
  const int half = lane >> 4;
  const int m16  = lane & 15;
  const int rowT = wave * 16 + m16;
  int node = blockIdx.x * TILE + rowT;
  if (node > NNODES - 1) node = NNODES - 1;          // duplicates write identical data

  if (wave == 0) {
    tdm_load_f32(params, Pbuf, 5 * DD);
    __builtin_amdgcn_s_wait_tensorcnt(0);
  } else if (wave == 1) {
#pragma unroll
    for (int i = 0; i < 4; ++i)
      __builtin_prefetch(w1t + ((size_t)(i * 32 + lane)) * 128, 0, 3);
  } else if (wave == 2) {
#pragma unroll
    for (int i = 0; i < 4; ++i)
      __builtin_prefetch(w2t + ((size_t)(i * 32 + lane)) * 128, 0, 3);
  }
  __syncthreads();

  const float* s0 = xbuf + (size_t)node * DD;
  const float* s1 = agg + (size_t)node * DD;

  v8f acc[8];
#pragma unroll
  for (int t = 0; t < 8; ++t) acc[t] = vzero8();

  // ---- Layer 1: [64,256] x [256,128] ----
  for (int kk = 0; kk < 8; ++kk) {
    const int kwin = kk * 32;
    const float* s = (kwin < DD) ? (s0 + kwin) : (s1 + (kwin - DD));
    v16bf b[8];
#pragma unroll
    for (int t = 0; t < 8; ++t)
      b[t] = bfrag(w0t + (size_t)(t * 16 + m16) * 256 + kwin + half * 8);
    v16bf a = afrag_f32(s + half * 8);
#pragma unroll
    for (int t = 0; t < 8; ++t) acc[t] = WMMA_BF16(a, b[t], acc[t]);
  }
  const int mbase = wave * 16 + half * 8;
  bias_relu_store(Hbuf, Pbuf, m16, mbase, acc);

  const bf16_t* arow = Hbuf + (size_t)rowT * HP;
  gemm128(arow, w1t, m16, half, acc);
  bias_relu_store(Hbuf, Pbuf + DD, m16, mbase, acc);
  gemm128(arow, w2t, m16, half, acc);
  __syncthreads();
  bias_store_f32(LNbuf, Pbuf + 2 * DD, m16, mbase, acc);

  const float* gam = Pbuf + 3 * DD;
  const float* bet = Pbuf + 4 * DD;
  const float* hrow = LNbuf + (size_t)rowT * LP;
  float sum = 0.0f, sq = 0.0f;
  for (int n4 = 0; n4 < DD; n4 += 4) {
    v4f h = *(const v4f*)(hrow + n4);
#pragma unroll
    for (int j = 0; j < 4; ++j) { sum += h[j]; sq += h[j] * h[j]; }
  }
  const float mu  = sum * (1.0f / DD);
  const float var = sq * (1.0f / DD) - mu * mu;
  const float rs  = rsqrtf(var + LN_EPS);

  float* xout = xbuf + (size_t)node * DD;
  const int nb = half * 64;
  for (int n4 = 0; n4 < 64; n4 += 4) {
    const int n = nb + n4;
    v4f h  = *(const v4f*)(hrow + n);
    v4f xo = *(const v4f*)(xout + n);
    v4f y;
#pragma unroll
    for (int j = 0; j < 4; ++j)
      y[j] = gam[n + j] * (h[j] - mu) * rs + bet[n + j] + xo[j];
    *(v4f*)(xout + n) = y;
  }
}

// ---------------------------------------------------------------------------
// Weight prep: f32 W[it][fi][fo] -> bf16 Wt[it][fo][fi]
// ---------------------------------------------------------------------------
__global__ void prep_weights(const float* __restrict__ eW0, const float* __restrict__ eW1,
                             const float* __restrict__ eW2, const float* __restrict__ nW0,
                             const float* __restrict__ nW1, const float* __restrict__ nW2,
                             bf16_t* __restrict__ w) {
  const int fis[6]  = {384, 128, 128, 256, 128, 128};
  const int offs[6] = {0, 49152, 65536, 81920, 114688, 131072};
  const int it = blockIdx.x / 6;
  const int ly = blockIdx.x % 6;
  const int fi = fis[ly];
  const float* srcs[6] = {eW0, eW1, eW2, nW0, nW1, nW2};
  const float* src = srcs[ly] + (size_t)it * fi * DD;
  bf16_t* dst = w + (size_t)it * 147456 + offs[ly];
  for (int i = threadIdx.x; i < fi * DD; i += blockDim.x) {
    const int n = i / fi;
    const int k = i - n * fi;
    dst[i] = (bf16_t)src[(size_t)k * DD + n];
  }
}

// Pack per-iteration params: [b0 | b1 | b2 | gamma | beta] (640 f32 each iter)
__global__ void prep_params(const float* __restrict__ b0, const float* __restrict__ b1,
                            const float* __restrict__ b2, const float* __restrict__ g,
                            const float* __restrict__ bt, float* __restrict__ dst) {
  const int it = blockIdx.x;
  const float* srcs[5] = {b0, b1, b2, g, bt};
  for (int i = threadIdx.x; i < 5 * DD; i += blockDim.x)
    dst[(size_t)it * 5 * DD + i] = srcs[i >> 7][(size_t)it * DD + (i & (DD - 1))];
}

extern "C" void kernel_launch(void* const* d_in, const int* in_sizes, int n_in,
                              void* d_out, int out_size, void* d_ws, size_t ws_size,
                              hipStream_t stream) {
  const float* x     = (const float*)d_in[0];
  const float* ea    = (const float*)d_in[1];
  const float* eW0   = (const float*)d_in[2];
  const float* eB0   = (const float*)d_in[3];
  const float* eW1   = (const float*)d_in[4];
  const float* eB1   = (const float*)d_in[5];
  const float* eW2   = (const float*)d_in[6];
  const float* eB2   = (const float*)d_in[7];
  const float* eG    = (const float*)d_in[8];
  const float* eBeta = (const float*)d_in[9];
  const float* nW0   = (const float*)d_in[10];
  const float* nB0   = (const float*)d_in[11];
  const float* nW1   = (const float*)d_in[12];
  const float* nB1   = (const float*)d_in[13];
  const float* nW2   = (const float*)d_in[14];
  const float* nB2   = (const float*)d_in[15];
  const float* nG    = (const float*)d_in[16];
  const float* nBeta = (const float*)d_in[17];
  const int*   eidx  = (const int*)d_in[18];

  char* ws = (char*)d_ws;
  const size_t xBytes = (size_t)NNODES * DD * sizeof(float);   // 10.24 MB
  const size_t eBytes = (size_t)NEDGES * DD * sizeof(float);   // 40.96 MB
  const size_t wBytes = (size_t)MPITERS * 147456 * sizeof(bf16_t);
  float*  xbuf = (float*)(ws);
  float*  ebuf = (float*)(ws + xBytes);
  float*  agg  = (float*)(ws + xBytes + eBytes);
  bf16_t* wbuf = (bf16_t*)(ws + 2 * xBytes + eBytes);
  float*  pE   = (float*)(ws + 2 * xBytes + eBytes + wBytes);
  float*  pN   = pE + (size_t)MPITERS * 5 * DD;

  hipMemcpyAsync(xbuf, x, xBytes, hipMemcpyDeviceToDevice, stream);
  hipMemcpyAsync(ebuf, ea, eBytes, hipMemcpyDeviceToDevice, stream);
  prep_weights<<<MPITERS * 6, 256, 0, stream>>>(eW0, eW1, eW2, nW0, nW1, nW2, wbuf);
  prep_params<<<MPITERS, 128, 0, stream>>>(eB0, eB1, eB2, eG, eBeta, pE);
  prep_params<<<MPITERS, 128, 0, stream>>>(nB0, nB1, nB2, nG, nBeta, pN);

  for (int it = 0; it < MPITERS; ++it) {
    hipMemsetAsync(agg, 0, xBytes, stream);
    const bf16_t* wi = wbuf + (size_t)it * 147456;
    edge_kernel<<<NEDGES / TILE, THREADS, 0, stream>>>(
        xbuf, ebuf, agg, eidx, wi, wi + 49152, wi + 65536, pE + (size_t)it * 5 * DD);
    node_kernel<<<(NNODES + TILE - 1) / TILE, THREADS, 0, stream>>>(
        xbuf, agg, wi + 81920, wi + 114688, wi + 131072, pN + (size_t)it * 5 * DD);
  }

  hipMemcpyAsync(d_out, xbuf, xBytes, hipMemcpyDeviceToDevice, stream);
  hipMemcpyAsync((float*)d_out + (size_t)NNODES * DD, ebuf, eBytes,
                 hipMemcpyDeviceToDevice, stream);
}
